// NystromAttention_73186242724039
// MI455X (gfx1250) — matrix operand order repaired and checked by reading, not compiled, not run on hardware
//
#include <hip/hip_runtime.h>
#include <hip/hip_bf16.h>
#include <cstdint>

// ---------------------------------------------------------------------------
// Nystrom attention for MI455X (gfx1250, wave32, WMMA bf16 16x16x32).
// All heavy math runs through v_wmma_f32_16x16x32_bf16 with fp32 accumulate.
// Each wave computes a 32x64 output tile: 8 WMMAs per 32-wide K step with
// 12 b128 loads (1.5 loads/WMMA) -> high L2 arithmetic intensity.
// ---------------------------------------------------------------------------

typedef __bf16 bf16_t;
typedef bf16_t v16bf __attribute__((ext_vector_type(16)));
typedef bf16_t v8bf  __attribute__((ext_vector_type(8)));
typedef float  v8f   __attribute__((ext_vector_type(8)));

#define B_     4
#define S_     4096
#define HID_   1024
#define NH_    16
#define HD_    64
#define M_     64
#define SCALE_ 0.125f          // 1/sqrt(64)
#define EPS_   1e-6f

// fp32 -> bf16, round-to-nearest-even, pure bit ops (no __bf16 arithmetic).
__device__ __forceinline__ bf16_t f2bf(float f) {
  unsigned u = __builtin_bit_cast(unsigned, f);
  unsigned r = u + 0x7FFFu + ((u >> 16) & 1u);
  unsigned short h = (unsigned short)(r >> 16);
  return __builtin_bit_cast(bf16_t, h);
}

enum Epi {
  EPI_QK_HEADS   = 0,  // (16384 x 1024) -> q/k [b][h][s][d] bf16
  EPI_VT_HEADS   = 1,  // (16384 x 1024) -> v^T [b][h][d][s] bf16
  EPI_PLAIN_BF16 = 2,  // C[z][m][n] bf16
  EPI_TRANS_BF16 = 3,  // C[z][n][m] bf16
  EPI_CTX        = 4,  // per-(b,h) (s x d) -> ctx [b][s][h*64+d] bf16
  EPI_PLAIN_F32  = 5   // C[z][m][n] fp32
};

// ---------------------------------------------------------------------------
// elementwise fp32 -> bf16 convert
// ---------------------------------------------------------------------------
__global__ void f32_to_bf16_kernel(const float* __restrict__ src,
                                   bf16_t* __restrict__ dst, long long n) {
  long long i = (long long)blockIdx.x * blockDim.x + threadIdx.x;
  long long stride = (long long)gridDim.x * blockDim.x;
  for (; i < n; i += stride) dst[i] = f2bf(src[i]);
}

// ---------------------------------------------------------------------------
// landmark gather: k_l[b][h][m][d] = k[b][h][65*m][d]
// ---------------------------------------------------------------------------
__global__ void gather_landmarks_kernel(const bf16_t* __restrict__ k,
                                        bf16_t* __restrict__ kl) {
  int i = blockIdx.x * blockDim.x + threadIdx.x;   // B*NH*M*HD = 262144
  if (i >= B_ * NH_ * M_ * HD_) return;
  int d  = i & (HD_ - 1);
  int m  = (i >> 6) & (M_ - 1);
  int bh = i >> 12;
  kl[i] = k[(long long)bh * S_ * HD_ + (long long)(65 * m) * HD_ + d];
}

// ---------------------------------------------------------------------------
// Generic batched GEMM: C = scale * A(MxK) * Bt(NxK)^T (+ bias[n]).
// One wave computes a 32x64 output tile: per 32-wide K step, 2 A fragments
// and 4 B fragments feed 8 v_wmma_f32_16x16x32_bf16. Here M%32==0, N%64==0,
// K%32==0 for every call.
// ---------------------------------------------------------------------------
__global__ __launch_bounds__(256) void gemm_bf16_wmma(
    const bf16_t* __restrict__ A, long long a_bstride, int lda,
    const bf16_t* __restrict__ Bt, long long b_bstride, int ldb,
    const float* __restrict__ bias, float scale,
    int Mdim, int Ndim, int Kdim,
    void* __restrict__ C, long long c_bstride, int ldc,
    int epi, int tiles) {
  const int lane = threadIdx.x & 31;
  const int wid  = threadIdx.x >> 5;
  const int gw   = blockIdx.x * 8 + wid;          // wave-uniform
  if (gw >= tiles) return;                        // EXEC stays all-ones per wave
  const int tiles_n = Ndim >> 6;
  const int tm = gw / tiles_n;                    // 32-row block
  const int tn = gw % tiles_n;                    // 64-col block
  const int z  = blockIdx.z;

  const bf16_t* Ab = A  + (long long)z * a_bstride;
  const bf16_t* Bb = Bt + (long long)z * b_bstride;

  const int half = lane >> 4;                     // K-half owned by this lane
  const int lmod = lane & 15;

  // A frags (16x32 bf16 each): lane = row; contiguous 16 K-values at half*16
  const bf16_t* arow0 = Ab + (long long)(tm * 32 + lmod) * lda + half * 16;
  const bf16_t* arow1 = arow0 + (long long)16 * lda;
  // B frags (32x16 bf16 each): lane = col; contiguous 16 K-values at half*16
  const bf16_t* brow0 = Bb + (long long)(tn * 64 + lmod) * ldb + half * 16;

  v8f acc00 = {}, acc01 = {}, acc02 = {}, acc03 = {};
  v8f acc10 = {}, acc11 = {}, acc12 = {}, acc13 = {};
  for (int k0 = 0; k0 < Kdim; k0 += 32) {
    union { v16bf v; v8bf h[2]; } a0, a1;
    a0.h[0] = *reinterpret_cast<const v8bf*>(arow0 + k0);
    a0.h[1] = *reinterpret_cast<const v8bf*>(arow0 + k0 + 8);
    a1.h[0] = *reinterpret_cast<const v8bf*>(arow1 + k0);
    a1.h[1] = *reinterpret_cast<const v8bf*>(arow1 + k0 + 8);
    v16bf b0 = *reinterpret_cast<const v16bf*>(brow0 + k0);
    v16bf b1 = *reinterpret_cast<const v16bf*>(brow0 + (long long)16 * ldb + k0);
    v16bf b2 = *reinterpret_cast<const v16bf*>(brow0 + (long long)32 * ldb + k0);
    v16bf b3 = *reinterpret_cast<const v16bf*>(brow0 + (long long)48 * ldb + k0);
    acc00 = __builtin_amdgcn_wmma_f32_16x16x32_bf16(false, a0.v, false, b0, (short)0, acc00, false, false);
    acc01 = __builtin_amdgcn_wmma_f32_16x16x32_bf16(false, a0.v, false, b1, (short)0, acc01, false, false);
    acc02 = __builtin_amdgcn_wmma_f32_16x16x32_bf16(false, a0.v, false, b2, (short)0, acc02, false, false);
    acc03 = __builtin_amdgcn_wmma_f32_16x16x32_bf16(false, a0.v, false, b3, (short)0, acc03, false, false);
    acc10 = __builtin_amdgcn_wmma_f32_16x16x32_bf16(false, a1.v, false, b0, (short)0, acc10, false, false);
    acc11 = __builtin_amdgcn_wmma_f32_16x16x32_bf16(false, a1.v, false, b1, (short)0, acc11, false, false);
    acc12 = __builtin_amdgcn_wmma_f32_16x16x32_bf16(false, a1.v, false, b2, (short)0, acc12, false, false);
    acc13 = __builtin_amdgcn_wmma_f32_16x16x32_bf16(false, a1.v, false, b3, (short)0, acc13, false, false);
  }

  v8f accs[2][4] = {{acc00, acc01, acc02, acc03}, {acc10, acc11, acc12, acc13}};
#pragma unroll
  for (int rb = 0; rb < 2; ++rb) {
#pragma unroll
    for (int t = 0; t < 4; ++t) {
      const int n = tn * 64 + t * 16 + lmod;
      const float bv = bias ? bias[n] : 0.0f;
#pragma unroll
      for (int j = 0; j < 8; ++j) {
        const int m = tm * 32 + rb * 16 + half * 8 + j;  // C layout: row = half*8+j
        const float v = accs[rb][t][j] * scale + bv;
        long long dst;
        switch (epi) {
          case EPI_QK_HEADS:
            dst = (long long)(m >> 12) * ((long long)NH_ * S_ * HD_) +
                  (long long)(n >> 6) * (S_ * HD_) +
                  (long long)(m & (S_ - 1)) * HD_ + (n & (HD_ - 1));
            ((bf16_t*)C)[dst] = f2bf(v);
            break;
          case EPI_VT_HEADS:
            dst = (long long)(m >> 12) * ((long long)NH_ * S_ * HD_) +
                  (long long)(n >> 6) * (S_ * HD_) +
                  (long long)(n & (HD_ - 1)) * S_ + (m & (S_ - 1));
            ((bf16_t*)C)[dst] = f2bf(v);
            break;
          case EPI_PLAIN_BF16:
            dst = (long long)z * c_bstride + (long long)m * ldc + n;
            ((bf16_t*)C)[dst] = f2bf(v);
            break;
          case EPI_TRANS_BF16:
            dst = (long long)z * c_bstride + (long long)n * ldc + m;
            ((bf16_t*)C)[dst] = f2bf(v);
            break;
          case EPI_CTX:
            dst = (long long)(z >> 4) * ((long long)S_ * HID_) +
                  (long long)m * HID_ + (long long)(z & 15) * HD_ + n;
            ((bf16_t*)C)[dst] = f2bf(v);
            break;
          default:  // EPI_PLAIN_F32
            dst = (long long)z * c_bstride + (long long)m * ldc + n;
            ((float*)C)[dst] = v;
            break;
        }
      }
    }
  }
}

// ---------------------------------------------------------------------------
// 64x64 SPD inverse (Gauss-Jordan, no pivoting needed: K_l K_l^T*s + eps*I).
// One block of 64 threads per matrix; thread t owns column t. Race-free:
// reads of shared pivot data happen before the sync that precedes writes.
// ---------------------------------------------------------------------------
__global__ __launch_bounds__(64) void invert64_kernel(const float* __restrict__ kk,
                                                      bf16_t* __restrict__ inv_bf) {
  __shared__ float a[64][65];
  __shared__ float w[64][65];
  __shared__ float fcol[64];
  const int z = blockIdx.x;
  const int t = threadIdx.x;
  const float* src = kk + (long long)z * 4096;
  for (int r = 0; r < 64; ++r) {
    a[r][t] = src[r * 64 + t] + ((r == t) ? EPS_ : 0.0f);
    w[r][t] = (r == t) ? 1.0f : 0.0f;
  }
  __syncthreads();
  for (int p = 0; p < 64; ++p) {
    const float pinv = 1.0f / a[p][p];
    const float fc = (t == p) ? 0.0f : a[t][p];
    __syncthreads();                 // all reads of row/col p done
    const float apc = a[p][t] * pinv; a[p][t] = apc;
    const float wpc = w[p][t] * pinv; w[p][t] = wpc;
    fcol[t] = fc;
    __syncthreads();                 // normalized row p + factors visible
    for (int r = 0; r < 64; ++r) {   // own-column updates only
      const float f = fcol[r];       // fcol[p]==0 -> row p untouched
      a[r][t] -= f * apc;
      w[r][t] -= f * wpc;
    }
    __syncthreads();
  }
  for (int r = 0; r < 64; ++r)
    inv_bf[(long long)z * 4096 + r * 64 + t] = f2bf(w[r][t]);
}

// ---------------------------------------------------------------------------
// host orchestration
// ---------------------------------------------------------------------------
extern "C" void kernel_launch(void* const* d_in, const int* in_sizes, int n_in,
                              void* d_out, int out_size, void* d_ws, size_t ws_size,
                              hipStream_t stream) {
  (void)in_sizes; (void)n_in; (void)out_size; (void)ws_size;
  const float* x  = (const float*)d_in[0];
  const float* Wq = (const float*)d_in[1];
  const float* bq = (const float*)d_in[2];
  const float* Wk = (const float*)d_in[3];
  const float* bk = (const float*)d_in[4];
  const float* Wv = (const float*)d_in[5];
  const float* bv = (const float*)d_in[6];
  const float* Wo = (const float*)d_in[7];
  const float* bo = (const float*)d_in[8];
  float* out = (float*)d_out;

  char* ws = (char*)d_ws;
  size_t off = 0;
  auto alloc = [&](size_t bytes) -> char* {
    char* p = ws + off;
    off = (off + bytes + 255) & ~(size_t)255;
    return p;
  };
  const size_t NTOK = (size_t)B_ * S_;                       // 16384
  bf16_t* x_bf  = (bf16_t*)alloc(NTOK * HID_ * 2);           // 32 MiB
  bf16_t* wq_bf = (bf16_t*)alloc((size_t)HID_ * HID_ * 2);
  bf16_t* wk_bf = (bf16_t*)alloc((size_t)HID_ * HID_ * 2);
  bf16_t* wv_bf = (bf16_t*)alloc((size_t)HID_ * HID_ * 2);
  bf16_t* wo_bf = (bf16_t*)alloc((size_t)HID_ * HID_ * 2);
  bf16_t* q_bf  = (bf16_t*)alloc(NTOK * HID_ * 2);           // [b][h][s][d]
  bf16_t* k_bf  = (bf16_t*)alloc(NTOK * HID_ * 2);           // [b][h][s][d]
  bf16_t* vt_bf = (bf16_t*)alloc(NTOK * HID_ * 2);           // [b][h][d][s]
  bf16_t* kl_bf = (bf16_t*)alloc((size_t)B_ * NH_ * M_ * HD_ * 2);
  float*  kk_f  = (float*) alloc((size_t)B_ * NH_ * M_ * M_ * 4);
  bf16_t* kinv  = (bf16_t*)alloc((size_t)B_ * NH_ * M_ * M_ * 2);
  bf16_t* qk_bf = (bf16_t*)alloc((size_t)B_ * NH_ * (size_t)S_ * M_ * 2);
  bf16_t* qs_bf = (bf16_t*)alloc((size_t)B_ * NH_ * (size_t)S_ * M_ * 2);
  bf16_t* klkvt = (bf16_t*)alloc((size_t)B_ * NH_ * HD_ * M_ * 2);
  bf16_t* klk_bf = x_bf;    // alias: x dead after V projection
  bf16_t* ctx_bf = vt_bf;   // alias: v^T dead after klkv

  auto gemm = [&](const bf16_t* Am, long long as, int lda,
                  const bf16_t* Bm, long long bs, int ldb,
                  const float* bias, float scale,
                  int Mdim, int Ndim, int Kdim,
                  void* Cm, long long cs, int ldc, int epi, int batches) {
    int tiles = (Mdim / 32) * (Ndim / 64);
    dim3 grid((unsigned)((tiles + 7) / 8), 1, (unsigned)batches);
    gemm_bf16_wmma<<<grid, 256, 0, stream>>>(Am, as, lda, Bm, bs, ldb, bias, scale,
                                             Mdim, Ndim, Kdim, Cm, cs, ldc, epi, tiles);
  };

  // 0) fp32 -> bf16 conversions
  f32_to_bf16_kernel<<<8192, 256, 0, stream>>>(x,  x_bf,  (long long)NTOK * HID_);
  f32_to_bf16_kernel<<<1024, 256, 0, stream>>>(Wq, wq_bf, (long long)HID_ * HID_);
  f32_to_bf16_kernel<<<1024, 256, 0, stream>>>(Wk, wk_bf, (long long)HID_ * HID_);
  f32_to_bf16_kernel<<<1024, 256, 0, stream>>>(Wv, wv_bf, (long long)HID_ * HID_);
  f32_to_bf16_kernel<<<1024, 256, 0, stream>>>(Wo, wo_bf, (long long)HID_ * HID_);

  // 1-3) q/k/v projections: x @ W^T + b, head-split (v stored transposed)
  gemm(x_bf, 0, HID_, wq_bf, 0, HID_, bq, 1.0f, (int)NTOK, HID_, HID_, q_bf,  0, 0, EPI_QK_HEADS, 1);
  gemm(x_bf, 0, HID_, wk_bf, 0, HID_, bk, 1.0f, (int)NTOK, HID_, HID_, k_bf,  0, 0, EPI_QK_HEADS, 1);
  gemm(x_bf, 0, HID_, wv_bf, 0, HID_, bv, 1.0f, (int)NTOK, HID_, HID_, vt_bf, 0, 0, EPI_VT_HEADS, 1);

  // 4) landmark gather k_l[b][h][m][d] = k[b][h][65m][d]
  gather_landmarks_kernel<<<(B_ * NH_ * M_ * HD_ + 255) / 256, 256, 0, stream>>>(k_bf, kl_bf);

  // 5) kk = SCALE * k_l k_l^T  (fp32 out, eps added inside inversion)
  gemm(kl_bf, 4096, HD_, kl_bf, 4096, HD_, nullptr, SCALE_, M_, M_, HD_, kk_f, 4096, M_, EPI_PLAIN_F32, B_ * NH_);

  // 6) kk_inv (symmetric -> usable directly as Bt)
  invert64_kernel<<<B_ * NH_, 64, 0, stream>>>(kk_f, kinv);

  // 7) qk = SCALE * q k_l^T
  gemm(q_bf, 262144, HD_, kl_bf, 4096, HD_, nullptr, SCALE_, S_, M_, HD_, qk_bf, 262144, M_, EPI_PLAIN_BF16, B_ * NH_);

  // 8) qk_scaled = qk @ kk_inv
  gemm(qk_bf, 262144, M_, kinv, 4096, M_, nullptr, 1.0f, S_, M_, M_, qs_bf, 262144, M_, EPI_PLAIN_BF16, B_ * NH_);

  // 9) klk = k_l k^T  ([b,h][m][s])
  gemm(kl_bf, 4096, HD_, k_bf, 262144, HD_, nullptr, 1.0f, M_, S_, HD_, klk_bf, 262144, S_, EPI_PLAIN_BF16, B_ * NH_);

  // 10) klkv = klk @ v  (stored transposed [b,h][d][m])
  gemm(klk_bf, 262144, S_, vt_bf, 262144, S_, nullptr, 1.0f, M_, HD_, S_, klkvt, 4096, M_, EPI_TRANS_BF16, B_ * NH_);

  // 11) ctx = qk_scaled @ klkv, merged back to [b][s][hid]
  gemm(qs_bf, 262144, M_, klkvt, 4096, M_, nullptr, 1.0f, S_, HD_, M_, ctx_bf, 0, 0, EPI_CTX, B_ * NH_);

  // 12) out = ctx @ Wo^T + bo  (fp32)
  gemm(ctx_bf, 0, HID_, wo_bf, 0, HID_, bo, 1.0f, (int)NTOK, HID_, HID_, out, 0, HID_, EPI_PLAIN_F32, 1);
}
